// InstanceBranch_6047313953220
// MI455X (gfx1250) — matrix operand order repaired and checked by reading, not compiled, not run on hardware
//
#include <hip/hip_runtime.h>

typedef __bf16 bf16_t;
typedef __attribute__((ext_vector_type(16))) __bf16 v16bf;
typedef __attribute__((ext_vector_type(8)))  __bf16 v8bf;
typedef __attribute__((ext_vector_type(8)))  float  v8f;
typedef int v4i __attribute__((vector_size(16)));

#define B_    8
#define C_    256
#define H_    128
#define W_    128
#define HW_   16384
#define NM    100      // num masks
#define NMP   112      // padded to 7*16
#define KD    128      // kernel dim
#define KC    2304     // 9*256 (K of implicit conv GEMM, ordered (r,s,c))
#define SLAB  (NMP * C_)   // 28672 bf16 = 57344 bytes per conv tap (rs)

#if defined(__has_builtin)
#  if __has_builtin(__builtin_amdgcn_global_load_async_to_lds_b128)
#    define HAVE_ASYNC_LDS 1
#  endif
#endif
#ifndef HAVE_ASYNC_LDS
#  define HAVE_ASYNC_LDS 0
#endif

// Stage one 57344-byte weight slab (wt2[rs]) into LDS: 256 threads x 14 x 16B.
__device__ __forceinline__ void stage_slab(bf16_t* dst, const bf16_t* src, int tid) {
#if HAVE_ASYNC_LDS
#pragma unroll
  for (int j = 0; j < 14; ++j) {
    const int q = j * 256 + tid;   // 16-byte chunk index, 3584 total
    __builtin_amdgcn_global_load_async_to_lds_b128(
        (__attribute__((address_space(1))) v4i*)(src + q * 8),
        (__attribute__((address_space(3))) v4i*)(dst + q * 8),
        0, 0);
  }
#else
#pragma unroll
  for (int j = 0; j < 14; ++j) {
    const int q = j * 256 + tid;
    *(v8bf*)(dst + q * 8) = *(const v8bf*)(src + q * 8);
  }
#endif
}

__device__ __forceinline__ void wait_stage() {
#if HAVE_ASYNC_LDS
#  if __has_builtin(__builtin_amdgcn_s_wait_asynccnt)
  __builtin_amdgcn_s_wait_asynccnt(0);
#  else
  asm volatile("s_wait_asynccnt 0" ::: "memory");
#  endif
#endif
}

// ---------------------------------------------------------------------------
// 1) fp32 features -> bf16 in two layouts:
//    feat_cp[b][c][p]  (for einsum B-matrix: K=p contiguous per channel)
//    feat_t [b][p][c]  (for conv A-matrix: 16 contiguous channels per lane)
// ---------------------------------------------------------------------------
__global__ void convert_features_k(const float* __restrict__ f,
                                   bf16_t* __restrict__ feat_cp,
                                   bf16_t* __restrict__ feat_t) {
  size_t i = (size_t)blockIdx.x * blockDim.x + threadIdx.x;   // B_*C_*HW_ exact
  int p = (int)(i % HW_);
  int c = (int)((i / HW_) % C_);
  int b = (int)(i / ((size_t)C_ * HW_));
  bf16_t bv = (bf16_t)f[i];
  feat_cp[i] = bv;
  feat_t[((size_t)b * HW_ + p) * C_ + c] = bv;
}

// ---------------------------------------------------------------------------
// 2) conv_w (100,256,3,3) fp32 -> wt2[rs][n][c] bf16, n padded to 112.
//    One contiguous 56KB slab per tap -> linear async staging into LDS.
// ---------------------------------------------------------------------------
__global__ void prep_weights_k(const float* __restrict__ w,
                               bf16_t* __restrict__ wt2) {
  int o = blockIdx.x * 256 + threadIdx.x;     // 9*NMP*C_ = 258048 exact
  int rs = o / SLAB;
  int rem = o % SLAB;
  int n = rem / C_;
  int c = rem % C_;
  int r = rs / 3, s = rs % 3;
  float v = 0.0f;
  if (n < NM) v = w[(((size_t)n * C_ + c) * 3 + r) * 3 + s];
  wt2[o] = (bf16_t)v;
}

// ---------------------------------------------------------------------------
// 3) Implicit-GEMM 3x3 conv via v_wmma_f32_16x16x32_bf16.
//    Block = 8 waves = one (b,h) output row (8 w-tiles of 16).
//    Weights double-buffered in LDS (2 x 56KB), staged with async-to-LDS;
//    all 8 waves share each slab -> 8x less weight traffic from L2.
//    K loop: 9 taps x (256 channels / 32). A fragment per lane = one spatial
//    position, 16 contiguous channels (two 16B loads); border -> zero frag.
// ---------------------------------------------------------------------------
__global__ __launch_bounds__(256) void conv_wmma_k(
    const bf16_t* __restrict__ feat_t,
    const bf16_t* __restrict__ wt2,
    const float* __restrict__ conv_b,
    float* __restrict__ iam) {
  __shared__ __align__(32) bf16_t smem[2 * SLAB];   // 112 KB

  const int b    = blockIdx.x >> 7;
  const int h    = blockIdx.x & 127;
  const int tid  = threadIdx.x;
  const int lane = tid & 31;
  const int wave = tid >> 5;
  const int half = lane >> 4;
  const int mcol = lane & 15;
  const int w0   = wave << 4;

  // Prefetch tap 0 slab into buffer 0.
  stage_slab(smem, wt2, tid);
  wait_stage();
  __syncthreads();

  v8f acc[7] = {};

  for (int rs = 0; rs < 9; ++rs) {
    const int cur = rs & 1;
    // Kick off staging of the next tap into the other buffer.
    if (rs + 1 < 9)
      stage_slab(smem + (1 - cur) * SLAB, wt2 + (size_t)(rs + 1) * SLAB, tid);

    const bf16_t* wbuf = smem + cur * SLAB;
    const int r = rs / 3, s = rs % 3;
    const int hh = h + r - 1;
    const int ww = w0 + mcol + s - 1;
    const bool valid = (hh >= 0) & (hh < H_) & (ww >= 0) & (ww < W_);
    const bf16_t* ap = feat_t + ((size_t)b * HW_ + (size_t)hh * W_ + ww) * C_;

    for (int c0 = 0; c0 < C_; c0 += 32) {
      v16bf a;
      if (valid) {
        const v8bf lo = *(const v8bf*)(ap + c0 + (half << 4));
        const v8bf hi = *(const v8bf*)(ap + c0 + 8 + (half << 4));
#pragma unroll
        for (int j = 0; j < 8; ++j) { a[j] = lo[j]; a[j + 8] = hi[j]; }
      } else {
#pragma unroll
        for (int j = 0; j < 16; ++j) a[j] = (bf16_t)0.0f;
      }
#pragma unroll
      for (int t = 0; t < 7; ++t) {
        const int ncol = (t << 4) + mcol;
        const v16bf bb =
            *(const v16bf*)(wbuf + (size_t)ncol * C_ + c0 + (half << 4));
        acc[t] = __builtin_amdgcn_wmma_f32_16x16x32_bf16(
            false, a, false, bb, (short)0, acc[t], false, false);
      }
    }
    // Ensure next slab landed (producer side) and all waves done reading
    // the buffer we are about to overwrite next iteration.
    wait_stage();
    __syncthreads();
  }

  // C layout: VGPR v -> M = v + 8*half, N = lane%16
#pragma unroll
  for (int t = 0; t < 7; ++t) {
    const int n = (t << 4) + mcol;
    if (n < NM) {
      const float cb = conv_b[n];
      float* orow = iam + (((size_t)b * NM + n) * H_ + h) * W_ + w0 + (half << 3);
#pragma unroll
      for (int v = 0; v < 8; ++v) orow[v] = acc[t][v] + cb;
    }
  }
}

// ---------------------------------------------------------------------------
// 4) Row softmax over HW=16384 per (b,n); writes bf16 prob (padded rows -> 0).
//    softmax_bias is a scalar added to every element -> shift-invariant no-op.
// ---------------------------------------------------------------------------
__global__ void softmax_k(const float* __restrict__ iam,
                          bf16_t* __restrict__ prob) {
  const int row = blockIdx.x;               // 0..B_*NMP-1
  const int b = row / NMP;
  const int n = row % NMP;
  const int tid = threadIdx.x;
  bf16_t* pr = prob + (size_t)row * HW_;
  if (n >= NM) {
    for (int p = tid; p < HW_; p += 256) pr[p] = (bf16_t)0.0f;
    return;
  }
  const float* src = iam + ((size_t)b * NM + n) * HW_;
  __shared__ float sred[256];
  float lmax = -3.402823466e38f;
  for (int p = tid; p < HW_; p += 256) lmax = fmaxf(lmax, src[p]);
  sred[tid] = lmax; __syncthreads();
  for (int s = 128; s > 0; s >>= 1) {
    if (tid < s) sred[tid] = fmaxf(sred[tid], sred[tid + s]);
    __syncthreads();
  }
  const float m = sred[0];
  __syncthreads();
  float lsum = 0.0f;
  for (int p = tid; p < HW_; p += 256) lsum += __expf(src[p] - m);
  sred[tid] = lsum; __syncthreads();
  for (int s = 128; s > 0; s >>= 1) {
    if (tid < s) sred[tid] += sred[tid + s];
    __syncthreads();
  }
  const float inv = 1.0f / sred[0];
  for (int p = tid; p < HW_; p += 256)
    pr[p] = (bf16_t)(__expf(src[p] - m) * inv);
}

// ---------------------------------------------------------------------------
// 5) inst[b,n,c] = sum_p prob[b,n,p]*feat[b,c,p] via WMMA bf16.
//    A row (prob) and B column (feat_cp channel) are both K-contiguous.
// ---------------------------------------------------------------------------
__global__ void einsum_wmma_k(const bf16_t* __restrict__ prob,
                              const bf16_t* __restrict__ featcp,
                              float* __restrict__ inst_raw) {
  const int blk = blockIdx.x;               // 8 * 7 * 16
  const int b   = blk / 112;
  const int rem = blk % 112;
  const int n0  = (rem / 16) << 4;
  const int c0  = (rem % 16) << 4;
  const int lane = threadIdx.x;
  const int half = lane >> 4;
  const int mcol = lane & 15;

  v8f acc = {};
  const bf16_t* aRow = prob   + ((size_t)b * NMP + n0 + mcol) * HW_;
  const bf16_t* bRow = featcp + ((size_t)b * C_  + c0 + mcol) * HW_;
  for (int k = 0; k < HW_; k += 32) {
    v16bf a;
    const v8bf lo = *(const v8bf*)(aRow + k + (half << 4));
    const v8bf hi = *(const v8bf*)(aRow + k + 8 + (half << 4));
#pragma unroll
    for (int j = 0; j < 8; ++j) { a[j] = lo[j]; a[j + 8] = hi[j]; }
    const v16bf bb = *(const v16bf*)(bRow + k + (half << 4));
    acc = __builtin_amdgcn_wmma_f32_16x16x32_bf16(
        false, a, false, bb, (short)0, acc, false, false);
  }
#pragma unroll
  for (int v = 0; v < 8; ++v) {
    const int n = n0 + v + (half << 3);
    inst_raw[((size_t)b * NMP + n) * C_ + c0 + mcol] = acc[v];
  }
}

// ---------------------------------------------------------------------------
// 6) fc (relu) + three heads, one block per (b,n). Tiny FLOPs -> scalar fp32.
// ---------------------------------------------------------------------------
__global__ void fc_heads_k(const float* __restrict__ inst_raw,
                           const float* __restrict__ fc_w,  const float* __restrict__ fc_b,
                           const float* __restrict__ cls_w, const float* __restrict__ cls_b,
                           const float* __restrict__ mk_w,  const float* __restrict__ mk_b,
                           const float* __restrict__ obj_w, const float* __restrict__ obj_b,
                           float* __restrict__ out_logits,
                           float* __restrict__ out_kernel,
                           float* __restrict__ out_scores) {
  const int row = blockIdx.x;               // b*NM + n
  const int b = row / NM;
  const int n = row % NM;
  const int tid = threadIdx.x;              // 256
  __shared__ float s_inst[C_];
  const float* iv = inst_raw + ((size_t)b * NMP + n) * C_;
  {
    float sum = fc_b[tid];
    const float* wrow = fc_w + (size_t)tid * C_;
    for (int k = 0; k < C_; ++k) sum += iv[k] * wrow[k];
    s_inst[tid] = fmaxf(sum, 0.0f);
  }
  __syncthreads();
  if (tid < KD) {
    float sum = mk_b[tid];
    const float* wrow = mk_w + (size_t)tid * C_;
    for (int c = 0; c < C_; ++c) sum += s_inst[c] * wrow[c];
    out_kernel[(size_t)row * KD + tid] = sum;
  } else if (tid == KD) {
    float sum = cls_b[0];
    for (int c = 0; c < C_; ++c) sum += s_inst[c] * cls_w[c];
    out_logits[row] = sum;
  } else if (tid == KD + 1) {
    float sum = obj_b[0];
    for (int c = 0; c < C_; ++c) sum += s_inst[c] * obj_w[c];
    out_scores[row] = sum;
  }
}

extern "C" void kernel_launch(void* const* d_in, const int* in_sizes, int n_in,
                              void* d_out, int out_size, void* d_ws, size_t ws_size,
                              hipStream_t stream) {
  (void)in_sizes; (void)n_in; (void)out_size; (void)ws_size;
  const float* features = (const float*)d_in[0];
  const float* conv_w   = (const float*)d_in[1];
  const float* conv_b   = (const float*)d_in[2];
  // d_in[3] = softmax_bias: scalar shift before softmax -> no-op
  const float* fc_w  = (const float*)d_in[4];
  const float* fc_b  = (const float*)d_in[5];
  const float* cls_w = (const float*)d_in[6];
  const float* cls_b = (const float*)d_in[7];
  const float* mk_w  = (const float*)d_in[8];
  const float* mk_b  = (const float*)d_in[9];
  const float* obj_w = (const float*)d_in[10];
  const float* obj_b = (const float*)d_in[11];

  float* out        = (float*)d_out;
  float* out_logits = out;                          // (8,100,1)
  float* out_kernel = out + 800;                    // (8,100,128)
  float* out_scores = out + 800 + 102400;           // (8,100,1)
  float* iam        = out + 104000;                 // (8,100,128,128)

  char* ws = (char*)d_ws;
  size_t off = 0;
  bf16_t* feat_cp = (bf16_t*)(ws + off); off += (size_t)B_ * C_ * HW_ * 2;   // 64 MB
  bf16_t* feat_t  = (bf16_t*)(ws + off); off += (size_t)B_ * C_ * HW_ * 2;   // 64 MB
  bf16_t* wt2     = (bf16_t*)(ws + off); off += (size_t)9 * SLAB * 2;        // 504 KB
  off = (off + 255) & ~(size_t)255;
  bf16_t* prob    = (bf16_t*)(ws + off); off += (size_t)B_ * NMP * HW_ * 2;  // 28 MB
  float*  inst_raw= (float*)(ws + off);                                      // 896 KB

  convert_features_k<<<131072, 256, 0, stream>>>(features, feat_cp, feat_t);
  prep_weights_k<<<1008, 256, 0, stream>>>(conv_w, wt2);
  conv_wmma_k<<<1024, 256, 0, stream>>>(feat_t, wt2, conv_b, iam);
  softmax_k<<<896, 256, 0, stream>>>(iam, prob);
  einsum_wmma_k<<<896, 32, 0, stream>>>(prob, feat_cp, inst_raw);
  fc_heads_k<<<800, 256, 0, stream>>>(inst_raw, fc_w, fc_b, cls_w, cls_b,
                                      mk_w, mk_b, obj_w, obj_b,
                                      out_logits, out_kernel, out_scores);
}
